// Beit3EncoderLayer_83726092468939
// MI455X (gfx1250) — matrix operand order repaired and checked by reading, not compiled, not run on hardware
//
#include <hip/hip_runtime.h>
#include <math.h>
#include <stdint.h>

typedef __attribute__((ext_vector_type(16))) __bf16 bf16x16;
typedef __attribute__((ext_vector_type(8)))  float  f32x8;

#define WMMA_BF16(a, b, c) \
  __builtin_amdgcn_wmma_f32_16x16x32_bf16(false, (a), false, (b), (short)0, (c), false, false)

constexpr int Bn = 8, Sn = 641, Dn = 1024, Hn = 16, FFn = 4096, SPn = 577, HDn = 64;

// ---------------------------------------------------------------------------
// CDNA5 async global->LDS staging (ASYNCcnt-tracked, ISA 10 / 15.18.3)
// ---------------------------------------------------------------------------
__device__ __forceinline__ void async_copy_b128(uint32_t lds_off, const float* gptr) {
  asm volatile("global_load_async_to_lds_b128 %0, %1, off"
               :
               : "v"(lds_off), "v"((uint64_t)(uintptr_t)gptr)
               : "memory");
}
template <int N>
__device__ __forceinline__ void async_wait() {
  asm volatile("s_wait_asynccnt %0" ::"i"(N) : "memory");
}
__device__ __forceinline__ uint32_t lds_offset(const void* p) {
  return (uint32_t)(uintptr_t)p;  // LDS aperture: addr[31:0] is the LDS offset
}

// ---------------------------------------------------------------------------
// Multiway LayerNorm: one block per token, NPT elements per thread (Dm=NPT*256)
// ---------------------------------------------------------------------------
template <int NPT>
__global__ __launch_bounds__(256) void mw_ln_kernel(
    const float* __restrict__ X, float* __restrict__ Y,
    const float* __restrict__ Gimg, const float* __restrict__ Bimg,
    const float* __restrict__ Gtxt, const float* __restrict__ Btxt) {
  constexpr int Dm = NPT * 256;
  const int tok = blockIdx.x;
  const int s = tok % Sn;
  const int st = (s < SPn) ? 0 : 1;
  const float* __restrict__ g  = st ? Gtxt : Gimg;
  const float* __restrict__ bb = st ? Btxt : Bimg;
  const float* __restrict__ x = X + (size_t)tok * Dm;
  float* __restrict__ y = Y + (size_t)tok * Dm;
  const int tid = threadIdx.x;

  float vals[NPT];
  float s1 = 0.f, s2 = 0.f;
#pragma unroll
  for (int i = 0; i < NPT; ++i) {
    const float v = x[tid + i * 256];
    vals[i] = v; s1 += v; s2 += v * v;
  }
#pragma unroll
  for (int off = 16; off > 0; off >>= 1) {
    s1 += __shfl_down(s1, off, 32);
    s2 += __shfl_down(s2, off, 32);
  }
  __shared__ float p1[8], p2[8];
  __shared__ float mv[2];
  const int wave = tid >> 5, lane = tid & 31;
  if (lane == 0) { p1[wave] = s1; p2[wave] = s2; }
  __syncthreads();
  if (tid == 0) {
    float a = 0.f, b2 = 0.f;
    for (int i = 0; i < 8; ++i) { a += p1[i]; b2 += p2[i]; }
    const float mean = a / (float)Dm;
    const float var = b2 / (float)Dm - mean * mean;
    mv[0] = mean;
    mv[1] = rsqrtf(var + 1e-5f);
  }
  __syncthreads();
  const float mean = mv[0], inv = mv[1];
#pragma unroll
  for (int i = 0; i < NPT; ++i) {
    const int c = tid + i * 256;
    y[c] = (vals[i] - mean) * inv * g[c] + bb[c];
  }
}

// ---------------------------------------------------------------------------
// Multiway GEMM, double-buffered async-LDS pipeline.
// Block tile BM=64 x BN=128, BK=32; 8 waves (2x4), each wave computes 32x32
// via 2 A-frags x 2 B-frags -> 4 accumulators -> 4 v_wmma per K-step.
// Stage i+1 async copies overlap stage i compute (s_wait_asynccnt 6 keeps the
// newest stage in flight). fp32 tiles in LDS; bf16 cvt at fragment build.
// Out-of-segment rows are clamped (their outputs are never stored).
// ---------------------------------------------------------------------------
template <int ACT>  // 0 = none, 1 = exact GELU
__global__ __launch_bounds__(256) void mw_gemm_kernel(
    const float* __restrict__ X, int ldx,
    const float* __restrict__ Wimg, const float* __restrict__ bimg,
    const float* __restrict__ Wtxt, const float* __restrict__ btxt,
    const float* __restrict__ Res,
    float* __restrict__ Y, int ldy, int K, int N, float scale) {
  const int seg = blockIdx.z;
  const int bat = seg >> 1, st = seg & 1;
  const int row0 = bat * Sn + (st ? SPn : 0);
  const int segRows = st ? (Sn - SPn) : SPn;
  const int mt = blockIdx.y;
  if (mt * 64 >= segRows) return;
  const float* __restrict__ W = st ? Wtxt : Wimg;
  const float* __restrict__ bias = st ? btxt : bimg;

  __shared__ float Asf[2][64][36];    // 2 x 64x32 fp32 (+pad, 16B-aligned rows)
  __shared__ float Bsf[2][32][132];   // 2 x 32x128 fp32 (+pad, 16B-aligned rows)

  const int tid = threadIdx.x;
  const int wave = tid >> 5, lane = tid & 31;
  const int wm = wave >> 2, wn = wave & 3;
  const int nBase = blockIdx.x * 128;

  f32x8 acc00 = {}, acc01 = {}, acc10 = {}, acc11 = {};

  // A loader: 2 x b128 per thread (64 rows x 8 chunks)
  const int ar0 = tid >> 3, ar1 = ar0 + 32, ac = (tid & 7) * 4;
  const int arc0 = (mt * 64 + ar0 < segRows) ? (mt * 64 + ar0) : (segRows - 1);
  const int arc1 = (mt * 64 + ar1 < segRows) ? (mt * 64 + ar1) : (segRows - 1);
  // B loader: 4 x b128 per thread (32 rows x 32 chunks)
  int brow[4], bcol[4];
#pragma unroll
  for (int i = 0; i < 4; ++i) {
    const int chunk = tid + i * 256;
    brow[i] = chunk >> 5;
    bcol[i] = (chunk & 31) * 4;
  }

  const int nSteps = K >> 5;
  auto issue = [&](int step) {
    const int stage = step & 1;
    const int k0 = step << 5;
    async_copy_b128(lds_offset(&Asf[stage][ar0][ac]),
                    X + (size_t)(row0 + arc0) * ldx + (k0 + ac));
    async_copy_b128(lds_offset(&Asf[stage][ar1][ac]),
                    X + (size_t)(row0 + arc1) * ldx + (k0 + ac));
#pragma unroll
    for (int i = 0; i < 4; ++i)
      async_copy_b128(lds_offset(&Bsf[stage][brow[i]][bcol[i]]),
                      W + (size_t)(k0 + brow[i]) * N + (nBase + bcol[i]));
  };

  issue(0);
  for (int s = 0; s < nSteps; ++s) {
    if (s + 1 < nSteps) {
      issue(s + 1);
      async_wait<6>();  // stage s complete; stage s+1 still in flight
    } else {
      async_wait<0>();
    }
    __syncthreads();

    const int stage = s & 1;
    const float (*As)[36]  = Asf[stage];
    const float (*Bs)[132] = Bsf[stage];

    bf16x16 a0, a1;
    {  // two A fragments (rows wm*32 .. +32)
      const int m = wm * 32 + (lane & 15);
      const int kb = (lane < 16) ? 0 : 16;
#pragma unroll
      for (int j = 0; j < 8; ++j) {
        a0[j]     = (__bf16)As[m][kb + j];
        a0[j + 8] = (__bf16)As[m][kb + 8 + j];
        a1[j]     = (__bf16)As[m + 16][kb + j];
        a1[j + 8] = (__bf16)As[m + 16][kb + 8 + j];
      }
    }
    bf16x16 b0, b1;
    {  // two B fragments (cols wn*32 .. +32)
      const int n = lane & 15;
      const int kb = (lane < 16) ? 0 : 16;
#pragma unroll
      for (int j = 0; j < 16; ++j) {
        b0[j] = (__bf16)Bs[kb + j][wn * 32 + n];
        b1[j] = (__bf16)Bs[kb + j][wn * 32 + 16 + n];
      }
    }
    acc00 = WMMA_BF16(a0, b0, acc00);
    acc01 = WMMA_BF16(a0, b1, acc01);
    acc10 = WMMA_BF16(a1, b0, acc10);
    acc11 = WMMA_BF16(a1, b1, acc11);
    __syncthreads();
  }

  // Epilogue: bias, scale, activation, residual
  const int n0 = nBase + wn * 32 + (lane & 15);
  const int n1 = n0 + 16;
  const float bv0 = bias[n0], bv1 = bias[n1];
  auto store_pair = [&](int grow, float v0, float v1) {
    if (grow >= segRows) return;
    const size_t orow = (size_t)(row0 + grow);
    v0 = (v0 + bv0) * scale;
    v1 = (v1 + bv1) * scale;
    if (ACT == 1) {
      v0 = 0.5f * v0 * (1.0f + erff(v0 * 0.70710678118654752f));
      v1 = 0.5f * v1 * (1.0f + erff(v1 * 0.70710678118654752f));
    }
    if (Res) {
      v0 += Res[orow * (size_t)ldy + n0];
      v1 += Res[orow * (size_t)ldy + n1];
    }
    Y[orow * (size_t)ldy + n0] = v0;
    Y[orow * (size_t)ldy + n1] = v1;
  };
#pragma unroll
  for (int r = 0; r < 8; ++r) {
    const int lrow = wm * 32 + ((lane < 16) ? r : r + 8);
    store_pair(mt * 64 + lrow,      acc00[r], acc01[r]);
    store_pair(mt * 64 + lrow + 16, acc10[r], acc11[r]);
  }
}

// ---------------------------------------------------------------------------
// Attention: one block per (batch, head, 16-row q-tile). Branchless masking:
// key mask precomputed to LDS (-inf for masked/out-of-range keys); addresses
// clamped in-bounds (garbage columns/rows are masked or never stored).
// ---------------------------------------------------------------------------
__global__ __launch_bounds__(256) void attn_kernel(
    const float* __restrict__ Q, const float* __restrict__ Kt,
    const float* __restrict__ V, const int* __restrict__ amask,
    const int* __restrict__ itm, float* __restrict__ Ctx) {
  const int bat = blockIdx.z, h = blockIdx.y, qt = blockIdx.x;
  const int q0 = qt * 16;

  __shared__ __bf16 Qs[16][72];    // q tile bf16 (+pad)
  __shared__ float  Sc[16][672];   // score rows (641 + pad to 21*32)
  __shared__ float  kmask[672];    // -inf for masked / OOR keys, else 0
  __shared__ float  Cx[16][64];    // partial context (upper key half)

  const int tid = threadIdx.x;
  const int wave = tid >> 5, lane = tid & 31;
  const int m = lane & 15;

  {  // load Q tile 16x64 -> bf16 (vectorized)
    const int r = tid >> 4, c = (tid & 15) * 4;
    const int qrow = (q0 + r < Sn) ? (q0 + r) : (Sn - 1);
    const float4 qv = *(const float4*)(Q + ((size_t)(bat * Sn + qrow)) * Dn + h * HDn + c);
    Qs[r][c + 0] = (__bf16)qv.x;
    Qs[r][c + 1] = (__bf16)qv.y;
    Qs[r][c + 2] = (__bf16)qv.z;
    Qs[r][c + 3] = (__bf16)qv.w;
  }
  // key mask (branchless epilogue later)
  for (int c = tid; c < 672; c += 256)
    kmask[c] = (c < Sn && amask[bat * Sn + c] != 0) ? 0.f : -INFINITY;
  {  // init never-written padding score columns
    const int r = tid >> 4, c = 656 + (tid & 15);
    Sc[r][c] = -INFINITY;
  }
  __syncthreads();

  // ---- Phase 1: scores = Q K^T (+ masks); waves stripe over 41 key tiles
  for (int kt = wave; kt < 41; kt += 8) {
    const int kb = kt * 16;
    bf16x16 aLo, aHi;
    {
      const int kbo = (lane < 16) ? 0 : 16;
#pragma unroll
      for (int j = 0; j < 8; ++j) {
        aLo[j]     = Qs[m][kbo + j];
        aLo[j + 8] = Qs[m][kbo + 8 + j];
        aHi[j]     = Qs[m][32 + kbo + j];
        aHi[j + 8] = Qs[m][40 + kbo + j];
      }
    }
    const int key = kb + (lane & 15);
    const int keyc = (key < Sn) ? key : (Sn - 1);  // clamp; OOR cols get -inf below
    bf16x16 bLo, bHi;
    {
      const int base = (lane < 16) ? 0 : 16;
      const float* kp = Kt + ((size_t)(bat * Sn + keyc)) * Dn + h * HDn;
#pragma unroll
      for (int j = 0; j < 16; ++j) {
        bLo[j] = (__bf16)kp[base + j];
        bHi[j] = (__bf16)kp[32 + base + j];
      }
    }
    f32x8 acc = {};
    acc = WMMA_BF16(aLo, bLo, acc);
    acc = WMMA_BF16(aHi, bHi, acc);
    const float km = kmask[key];
#pragma unroll
    for (int r = 0; r < 8; ++r) {
      const int lrow = (lane < 16) ? r : r + 8;
      const int qrowc = (q0 + lrow < Sn) ? (q0 + lrow) : (Sn - 1);
      const float ib = (itm[(size_t)qrowc * Sn + keyc] != 0) ? -1e8f : 0.f;
      Sc[lrow][key] = acc[r] + ib + km;
    }
  }
  __syncthreads();

  // ---- Phase 2: softmax in place, one wave per 2 rows
#pragma unroll
  for (int rr = 0; rr < 2; ++rr) {
    const int row = wave * 2 + rr;
    float mx = -INFINITY;
    for (int c = lane; c < 672; c += 32) mx = fmaxf(mx, Sc[row][c]);
#pragma unroll
    for (int off = 16; off > 0; off >>= 1) mx = fmaxf(mx, __shfl_down(mx, off, 32));
    mx = __shfl(mx, 0, 32);
    float sum = 0.f;
    for (int c = lane; c < 672; c += 32) sum += expf(Sc[row][c] - mx);
#pragma unroll
    for (int off = 16; off > 0; off >>= 1) sum += __shfl_down(sum, off, 32);
    sum = __shfl(sum, 0, 32);
    const float inv = 1.0f / sum;
    for (int c = lane; c < 672; c += 32) Sc[row][c] = expf(Sc[row][c] - mx) * inv;
  }
  __syncthreads();

  // ---- Phase 3: ctx = P @ V; wave -> (nsub 0..3, key-half 0..1)
  const int nsub = wave & 3, half = wave >> 2;
  f32x8 acc = {};
  const int t0 = half ? 11 : 0, t1 = half ? 21 : 11;
  for (int t = t0; t < t1; ++t) {
    const int kb = t * 32;
    bf16x16 a;
    {
      const int kbo = (lane < 16) ? 0 : 16;
#pragma unroll
      for (int j = 0; j < 8; ++j) {
        a[j]     = (__bf16)Sc[m][kb + kbo + j];
        a[j + 8] = (__bf16)Sc[m][kb + 8 + kbo + j];
      }
    }
    bf16x16 bfr;
    {
      const int dcol = h * HDn + nsub * 16 + (lane & 15);
      const int base = (lane < 16) ? 0 : 16;
#pragma unroll
      for (int j = 0; j < 16; ++j) {
        const int key = kb + base + j;
        const int keyc = (key < Sn) ? key : (Sn - 1);  // probs are 0 for OOR keys
        bfr[j] = (__bf16)V[((size_t)(bat * Sn + keyc)) * Dn + dcol];
      }
    }
    acc = WMMA_BF16(a, bfr, acc);
  }
  const int col = nsub * 16 + (lane & 15);
  if (half == 1) {
#pragma unroll
    for (int r = 0; r < 8; ++r) {
      const int lrow = (lane < 16) ? r : r + 8;
      Cx[lrow][col] = acc[r];
    }
  }
  __syncthreads();
  if (half == 0) {
#pragma unroll
    for (int r = 0; r < 8; ++r) {
      const int lrow = (lane < 16) ? r : r + 8;
      const int qrow = q0 + lrow;
      if (qrow < Sn)
        Ctx[((size_t)(bat * Sn + qrow)) * Dn + h * HDn + col] = acc[r] + Cx[lrow][col];
    }
  }
}

// ---------------------------------------------------------------------------
extern "C" void kernel_launch(void* const* d_in, const int* in_sizes, int n_in,
                              void* d_out, int out_size, void* d_ws, size_t ws_size,
                              hipStream_t stream) {
  (void)in_sizes; (void)n_in; (void)out_size; (void)ws_size;
  const float* hidden = (const float*)d_in[0];
  const int*   amask  = (const int*)d_in[1];
  const int*   itm    = (const int*)d_in[2];
  // d_in[3] = multiway_split_position (compile-time constant SPn)
  const float* P[40];
  for (int i = 0; i < 40; ++i) P[i] = (const float*)d_in[4 + i];
  // P index map (d_in[4+i]):
  //  0..3  q_{img,txt}_{w,b}    4..7  k_*    8..11 v_*   12..15 out_*
  // 16..19 attn_ln_{img,txt}_{g,b}  20..23 inner_ln_*  24..27 final_ln_*
  // 28..33 ffn_img: fc1_w, fc1_b, fc2_w, fc2_b, ln_g, ln_b
  // 34..39 ffn_txt: fc1_w, fc1_b, fc2_w, fc2_b, ln_g, ln_b

  float* ws = (float*)d_ws;
  const size_t ND = (size_t)Bn * Sn * Dn;
  float* Abuf = ws;            // xln / context (reused)
  float* Qb   = ws + ND;       // q / inner_ln out (reused)
  float* Kb   = ws + 2 * ND;   // k / post-attn x (reused)
  float* Vb   = ws + 3 * ND;   // v / final_ln out (reused)
  float* Hb   = ws + 4 * ND;   // FFN hidden (B*S*FF)
  float* out  = (float*)d_out;

  const dim3 blk(256);
  const dim3 gD(8, 10, 2 * Bn);    // N=1024 GEMMs, BM=64 -> ceil(577/64)=10
  const dim3 gF1(32, 10, 2 * Bn);  // N=4096 GEMM
  const dim3 gA(41, Hn, Bn);       // attention

  // 1) pre-norm
  mw_ln_kernel<4><<<Bn * Sn, blk, 0, stream>>>(hidden, Abuf, P[16], P[17], P[18], P[19]);
  // 2) Q/K/V projections (q scaled by HD^-0.5)
  mw_gemm_kernel<0><<<gD, blk, 0, stream>>>(Abuf, Dn, P[0], P[1], P[2], P[3], nullptr, Qb, Dn, Dn, Dn, 0.125f);
  mw_gemm_kernel<0><<<gD, blk, 0, stream>>>(Abuf, Dn, P[4], P[5], P[6], P[7], nullptr, Kb, Dn, Dn, Dn, 1.0f);
  mw_gemm_kernel<0><<<gD, blk, 0, stream>>>(Abuf, Dn, P[8], P[9], P[10], P[11], nullptr, Vb, Dn, Dn, Dn, 1.0f);
  // 3) attention -> Abuf
  attn_kernel<<<gA, blk, 0, stream>>>(Qb, Kb, Vb, amask, itm, Abuf);
  // 4) inner LN -> Qb
  mw_ln_kernel<4><<<Bn * Sn, blk, 0, stream>>>(Abuf, Qb, P[20], P[21], P[22], P[23]);
  // 5) out projection + residual(hidden) -> Kb
  mw_gemm_kernel<0><<<gD, blk, 0, stream>>>(Qb, Dn, P[12], P[13], P[14], P[15], hidden, Kb, Dn, Dn, Dn, 1.0f);
  // 6) final LN -> Vb
  mw_ln_kernel<4><<<Bn * Sn, blk, 0, stream>>>(Kb, Vb, P[24], P[25], P[26], P[27]);
  // 7) fc1 + exact GELU -> Hb
  mw_gemm_kernel<1><<<gF1, blk, 0, stream>>>(Vb, Dn, P[28], P[29], P[34], P[35], nullptr, Hb, FFn, Dn, FFn, 1.0f);
  // 8) FFN sub-LN in place
  mw_ln_kernel<16><<<Bn * Sn, blk, 0, stream>>>(Hb, Hb, P[32], P[33], P[38], P[39]);
  // 9) fc2 + residual(Kb) -> out
  mw_gemm_kernel<0><<<gD, blk, 0, stream>>>(Hb, FFn, P[30], P[31], P[36], P[37], Kb, out, Dn, FFn, Dn, 1.0f);
}
